// SparseAttention_6932077215900
// MI455X (gfx1250) — matrix operand order repaired and checked by reading, not compiled, not run on hardware
//
#include <hip/hip_runtime.h>

// Sliding-window attention (W=256 one-sided), B=2, S=4096, H=16, D=64, f32 I/O.
// bf16 WMMA flash-attention. K/V pre-packed to bf16 in d_ws:
//   Kp[b,h, s+W, d]   (row-major, zero-padded W rows each side of s)
//   Vp[b,h, d, s+W]   (transposed, zero-padded W cols each side of s)
// Main kernel: one wave32 per 16-query tile, 9 key tiles of 64 keys.
// Wave-uniform scalars forced into SGPRs via readfirstlane so all tile
// control flow is scalar branches (EXEC stays all-ones for WMMA).

#define S_LEN   4096
#define NHEADS  16
#define DHEAD   64
#define HID     1024
#define WIN     256
#define SP      (S_LEN + 2 * WIN)      // 4608 padded sequence length
#define NEG_BIG (-1.0e30f)

typedef __attribute__((ext_vector_type(16))) __bf16 bf16x16;
typedef __attribute__((ext_vector_type(8)))  __bf16 bf16x8;
typedef __attribute__((ext_vector_type(4)))  __bf16 bf16x4;
typedef __attribute__((ext_vector_type(8)))  float  f32x8;
typedef __attribute__((ext_vector_type(4)))  float  f32x4;

__device__ __forceinline__ float half_reduce_max(float x) {
  x = fmaxf(x, __shfl_xor(x, 1, 32));
  x = fmaxf(x, __shfl_xor(x, 2, 32));
  x = fmaxf(x, __shfl_xor(x, 4, 32));
  x = fmaxf(x, __shfl_xor(x, 8, 32));
  return x;
}
__device__ __forceinline__ float half_reduce_sum(float x) {
  x += __shfl_xor(x, 1, 32);
  x += __shfl_xor(x, 2, 32);
  x += __shfl_xor(x, 4, 32);
  x += __shfl_xor(x, 8, 32);
  return x;
}

// ---------------- pad zeroing (Kp / Vp borders) ----------------
__global__ __launch_bounds__(256, 1)
void zero_pads(__bf16* __restrict__ Kp, __bf16* __restrict__ Vp, int BH) {
  const int i = blockIdx.x * 256 + threadIdx.x;
  const int kChunksPerBlk = WIN * DHEAD / 8;              // 2048 (16B chunks)
  const int nKchunks = BH * 2 * kChunksPerBlk;
  const bf16x8 z = {};
  if (i < nKchunks) {
    const int per = 2 * kChunksPerBlk;
    const int bh = i / per, r = i % per;
    const int blk = r / kChunksPerBlk, off = (r % kChunksPerBlk) * 8;
    const size_t base = (size_t)bh * SP * DHEAD + (size_t)blk * (WIN + S_LEN) * DHEAD + off;
    *(bf16x8*)(Kp + base) = z;
  } else {
    const int j = i - nKchunks;
    const int per = 2 * (WIN / 8);                        // 64 chunks per (bh,d) row
    const int row = j / per, r = j % per;
    const int blk = r / (WIN / 8), off = (r % (WIN / 8)) * 8;
    if (row < BH * DHEAD) {
      const size_t base = (size_t)row * SP + (size_t)blk * (WIN + S_LEN) + off;
      *(bf16x8*)(Vp + base) = z;
    }
  }
}

// ---------------- pack f32 K/V -> bf16 Kp (direct) / Vp (LDS transpose) ----------------
__global__ __launch_bounds__(256, 1)
void pack_kv(const float* __restrict__ K, const float* __restrict__ V,
             __bf16* __restrict__ Kp, __bf16* __restrict__ Vp) {
  __shared__ __bf16 tile[64 * 68];                        // [s][d], padded stride
  const int t  = threadIdx.x;
  const int bh = blockIdx.x >> 6;                         // S/64 = 64 chunks
  const int s0 = (blockIdx.x & 63) * 64;
  const int b  = bh >> 4;
  const int h  = bh & 15;
  const int c4 = (t & 15) * 4;
  #pragma unroll
  for (int it = 0; it < 4; ++it) {
    const int row = it * 16 + (t >> 4);
    const size_t src = ((size_t)b * S_LEN + s0 + row) * HID + h * DHEAD + c4;
    const f32x4 k4 = *(const f32x4*)(K + src);
    const f32x4 v4 = *(const f32x4*)(V + src);
    bf16x4 kb, vb;
    #pragma unroll
    for (int e = 0; e < 4; ++e) { kb[e] = (__bf16)k4[e]; vb[e] = (__bf16)v4[e]; }
    *(bf16x4*)(Kp + ((size_t)bh * SP + s0 + row + WIN) * DHEAD + c4) = kb;
    *(bf16x4*)(&tile[row * 68 + c4]) = vb;
  }
  __syncthreads();
  const int d    = t >> 2;
  const int sblk = (t & 3) * 16;
  bf16x16 o16;
  #pragma unroll
  for (int e = 0; e < 16; ++e) o16[e] = tile[(sblk + e) * 68 + d];
  *(bf16x16*)(Vp + ((size_t)bh * DHEAD + d) * SP + WIN + s0 + sblk) = o16;
}

// ---------------- main attention kernel (packed bf16 K/V, 64-key tiles) ----------------
__global__ __launch_bounds__(256, 1)
void swa_wmma_main(const float* __restrict__ Q, const __bf16* __restrict__ Kp,
                   const __bf16* __restrict__ Vp, float* __restrict__ out) {
  __shared__ __bf16 lds_p[8 * 16 * 64];                   // per-wave 16x64 P staging
  const int lane = threadIdx.x & 31;
  const int wv   = threadIdx.x >> 5;
  // wave-uniform -> SGPRs (scalar control flow, SGPR base addresses)
  const int gw   = __builtin_amdgcn_readfirstlane(blockIdx.x * 8 + wv);
  const int ntq  = S_LEN / 16;
  const int qt   = gw % ntq;
  const int h    = (gw / ntq) % NHEADS;
  const int b    = gw / (ntq * NHEADS);
  const int q0   = qt * 16;
  const int col  = lane & 15;
  const int hi   = lane >> 4;

  const size_t bhQ = (size_t)b * S_LEN * HID + (size_t)h * DHEAD;
  const size_t bhK = (size_t)(b * NHEADS + h) * SP * DHEAD;
  const size_t bhV = (size_t)(b * NHEADS + h) * DHEAD * SP;

  // Q A-fragments (contiguous k = 16*hi + e layout), loaded once from f32
  bf16x16 aq0, aq1;
  {
    const float* qrow = Q + bhQ + (size_t)(q0 + col) * HID;
    #pragma unroll
    for (int e = 0; e < 16; ++e) {
      const int kk = (hi ? 16 : 0) + e;
      aq0[e] = (__bf16)qrow[kk];
      aq1[e] = (__bf16)qrow[32 + kk];
    }
  }
  bf16x16 bones;                                          // all-ones B for row sums
  #pragma unroll
  for (int e = 0; e < 16; ++e) bones[e] = (__bf16)1.0f;

  const f32x8 fzero = {0.f, 0.f, 0.f, 0.f, 0.f, 0.f, 0.f, 0.f};
  f32x8 acc[4];
  #pragma unroll
  for (int i = 0; i < 4; ++i) acc[i] = fzero;
  float m_r[8], l_r[8];
  #pragma unroll
  for (int r = 0; r < 8; ++r) { m_r[r] = NEG_BIG; l_r[r] = 0.0f; }

  __bf16* Pld = lds_p + wv * (16 * 64);
  const int kstart = q0 - WIN;
  const int neg    = -(kstart + 63);
  const int t_lo   = neg > 0 ? (neg + 63) >> 6 : 0;       // first tile with a key >= 0
  const int t_hi   = min(8, (S_LEN - 1 - kstart) >> 6);   // last tile with a key < S

  #pragma unroll 1
  for (int t = t_lo; t <= t_hi; ++t) {
    const int kbase = kstart + t * 64;
    const int kp0   = kbase + WIN;                        // padded base, always valid

    // ---- scores: S(16x64) = Q @ K^T, 8 WMMAs over 4 key quarters ----
    f32x8 s[4];
    #pragma unroll
    for (int qq = 0; qq < 4; ++qq) {
      const __bf16* kr = Kp + bhK + (size_t)(kp0 + qq * 16 + col) * DHEAD + (hi ? 16 : 0);
      if (qq == 0) __builtin_prefetch((const void*)(kr + 64 * DHEAD), 0, 1);
      bf16x16 bk0, bk1;
      #pragma unroll
      for (int e = 0; e < 16; ++e) { bk0[e] = kr[e]; bk1[e] = kr[32 + e]; }
      f32x8 sv = __builtin_amdgcn_wmma_f32_16x16x32_bf16(false, aq0, false, bk0, (short)0, fzero, false, false);
      s[qq]    = __builtin_amdgcn_wmma_f32_16x16x32_bf16(false, aq1, false, bk1, (short)0, sv, false, false);
    }

    // ---- masking only on edge tiles (scalar branch) ----
    if ((t == 0) | (t == 8) | (kbase < 0) | (kbase + 63 >= S_LEN)) {
      #pragma unroll
      for (int qq = 0; qq < 4; ++qq) {
        #pragma unroll
        for (int r = 0; r < 8; ++r) {
          const int k   = kbase + qq * 16 + col;
          const int rel = k - (q0 + r + hi * 8);
          if (!(rel >= -WIN && rel <= WIN && k >= 0 && k < S_LEN)) s[qq][r] = NEG_BIG;
        }
      }
    }

    // ---- online softmax; convert+stage P into LDS as it is produced ----
    #pragma unroll
    for (int r = 0; r < 8; ++r) {
      float tm = fmaxf(fmaxf(s[0][r], s[1][r]), fmaxf(s[2][r], s[3][r]));
      tm = half_reduce_max(tm);
      const float mnew  = fmaxf(m_r[r], tm);
      const float scale = __expf(m_r[r] - mnew);
      m_r[r]  = mnew;
      l_r[r] *= scale;
      acc[0][r] *= scale; acc[1][r] *= scale;
      acc[2][r] *= scale; acc[3][r] *= scale;
      __bf16* prow = Pld + (r + hi * 8) * 64 + col;
      #pragma unroll
      for (int qq = 0; qq < 4; ++qq)
        prow[qq * 16] = (__bf16)__expf(s[qq][r] - mnew);
    }
    __builtin_amdgcn_wave_barrier();
    asm volatile("s_wait_dscnt 0" ::: "memory");
    __builtin_amdgcn_wave_barrier();

    // reload P as two A-fragments (keys 0..31 / 32..63)
    bf16x16 ap0, ap1;
    #pragma unroll
    for (int e = 0; e < 16; ++e) {
      const int kk = (hi ? 16 : 0) + e;
      ap0[e] = Pld[col * 64 + kk];
      ap1[e] = Pld[col * 64 + 32 + kk];
    }

    // ---- row sums via WMMA against ones (keeps num/denom consistent) ----
    f32x8 rs = __builtin_amdgcn_wmma_f32_16x16x32_bf16(false, ap0, false, bones, (short)0, fzero, false, false);
    rs       = __builtin_amdgcn_wmma_f32_16x16x32_bf16(false, ap1, false, bones, (short)0, rs, false, false);
    #pragma unroll
    for (int r = 0; r < 8; ++r) l_r[r] += rs[r];

    // ---- O += P @ V: 8 WMMAs; transposed padded bf16 V, contiguous loads ----
    const size_t vb = bhV + (size_t)(kp0 + (hi ? 16 : 0));
    __builtin_prefetch((const void*)(Vp + vb + (size_t)col * SP + 64), 0, 1);
    #pragma unroll
    for (int j = 0; j < 4; ++j) {
      const __bf16* vr = Vp + vb + (size_t)(j * 16 + col) * SP;
      bf16x16 bv0, bv1;
      #pragma unroll
      for (int e = 0; e < 16; ++e) { bv0[e] = vr[e]; bv1[e] = vr[32 + e]; }
      acc[j] = __builtin_amdgcn_wmma_f32_16x16x32_bf16(false, ap0, false, bv0, (short)0, acc[j], false, false);
      acc[j] = __builtin_amdgcn_wmma_f32_16x16x32_bf16(false, ap1, false, bv1, (short)0, acc[j], false, false);
    }
  }

  // ---- normalize + store ----
  #pragma unroll
  for (int r = 0; r < 8; ++r) {
    const float inv = 1.0f / l_r[r];
    const size_t orow = ((size_t)b * S_LEN + q0 + r + hi * 8) * HID + (size_t)h * DHEAD + col;
    #pragma unroll
    for (int j = 0; j < 4; ++j) out[orow + j * 16] = acc[j][r] * inv;
  }
}

// ---------------- fallback: self-contained f32->bf16 inline path ----------------
__global__ __launch_bounds__(256, 1)
void swa_wmma_fallback(const float* __restrict__ Q, const float* __restrict__ K,
                       const float* __restrict__ V, float* __restrict__ out) {
  __shared__ __bf16 lds_p[8 * 16 * 32];
  const int lane = threadIdx.x & 31;
  const int wv   = threadIdx.x >> 5;
  const int gw   = __builtin_amdgcn_readfirstlane(blockIdx.x * 8 + wv);
  const int ntq  = S_LEN / 16;
  const int qt = gw % ntq;
  const int h  = (gw / ntq) % NHEADS;
  const int b  = gw / (ntq * NHEADS);
  const int q0 = qt * 16;
  const int col = lane & 15;
  const int hi  = lane >> 4;
  const size_t bh = (size_t)b * S_LEN * HID + (size_t)h * DHEAD;

  bf16x16 aq0, aq1;
  {
    const float* qrow = Q + bh + (size_t)(q0 + col) * HID;
    #pragma unroll
    for (int e = 0; e < 16; ++e) {
      const int kk = (hi ? 16 : 0) + e;
      aq0[e] = (__bf16)qrow[kk];
      aq1[e] = (__bf16)qrow[32 + kk];
    }
  }
  const f32x8 fzero = {0.f, 0.f, 0.f, 0.f, 0.f, 0.f, 0.f, 0.f};
  f32x8 acc[4];
  #pragma unroll
  for (int i = 0; i < 4; ++i) acc[i] = fzero;
  float m_r[8], l_r[8];
  #pragma unroll
  for (int r = 0; r < 8; ++r) { m_r[r] = NEG_BIG; l_r[r] = 0.0f; }
  __bf16* Pld = lds_p + wv * (16 * 32);
  const int kstart = q0 - WIN;

  #pragma unroll 1
  for (int t = 0; t < 17; ++t) {
    const int kbase = kstart + t * 32;
    if ((kbase + 31) < 0 || kbase >= S_LEN) continue;
    f32x8 s0 = fzero, s1 = fzero;
    {
      const int ck0 = min(max(kbase + col, 0), S_LEN - 1);
      const int ck1 = min(max(kbase + 16 + col, 0), S_LEN - 1);
      const float* kr0 = K + bh + (size_t)ck0 * HID;
      const float* kr1 = K + bh + (size_t)ck1 * HID;
      const int dbase = hi ? 16 : 0;
      bf16x16 b00, b01, b10, b11;
      #pragma unroll
      for (int e = 0; e < 16; ++e) {
        b00[e] = (__bf16)kr0[dbase + e];
        b01[e] = (__bf16)kr0[32 + dbase + e];
        b10[e] = (__bf16)kr1[dbase + e];
        b11[e] = (__bf16)kr1[32 + dbase + e];
      }
      s0 = __builtin_amdgcn_wmma_f32_16x16x32_bf16(false, aq0, false, b00, (short)0, s0, false, false);
      s0 = __builtin_amdgcn_wmma_f32_16x16x32_bf16(false, aq1, false, b01, (short)0, s0, false, false);
      s1 = __builtin_amdgcn_wmma_f32_16x16x32_bf16(false, aq0, false, b10, (short)0, s1, false, false);
      s1 = __builtin_amdgcn_wmma_f32_16x16x32_bf16(false, aq1, false, b11, (short)0, s1, false, false);
    }
    float p0[8], p1[8];
    #pragma unroll
    for (int r = 0; r < 8; ++r) {
      const int qrow = q0 + r + hi * 8;
      const int k0 = kbase + col, k1 = k0 + 16;
      const int r0 = k0 - qrow, r1 = k1 - qrow;
      float sv0 = s0[r], sv1 = s1[r];
      if (!(r0 >= -WIN && r0 <= WIN && k0 >= 0 && k0 < S_LEN)) sv0 = NEG_BIG;
      if (!(r1 >= -WIN && r1 <= WIN && k1 >= 0 && k1 < S_LEN)) sv1 = NEG_BIG;
      const float tmax  = half_reduce_max(fmaxf(sv0, sv1));
      const float mnew  = fmaxf(m_r[r], tmax);
      const float scale = __expf(m_r[r] - mnew);
      p0[r] = __expf(sv0 - mnew);
      p1[r] = __expf(sv1 - mnew);
      const float rsum = half_reduce_sum(p0[r] + p1[r]);
      l_r[r] = l_r[r] * scale + rsum;
      m_r[r] = mnew;
      acc[0][r] *= scale; acc[1][r] *= scale;
      acc[2][r] *= scale; acc[3][r] *= scale;
    }
    #pragma unroll
    for (int r = 0; r < 8; ++r) {
      Pld[(r + hi * 8) * 32 + col]      = (__bf16)p0[r];
      Pld[(r + hi * 8) * 32 + 16 + col] = (__bf16)p1[r];
    }
    __builtin_amdgcn_wave_barrier();
    asm volatile("s_wait_dscnt 0" ::: "memory");
    __builtin_amdgcn_wave_barrier();
    bf16x16 ap;
    #pragma unroll
    for (int e = 0; e < 16; ++e) {
      const int kk = (hi ? 16 : 0) + e;
      ap[e] = Pld[col * 32 + kk];
    }
    const int keyb = kbase + (hi ? 16 : 0);
    #pragma unroll
    for (int j = 0; j < 4; ++j) {
      bf16x16 bv;
      #pragma unroll
      for (int e = 0; e < 16; ++e) {
        const int ky = min(max(keyb + e, 0), S_LEN - 1);
        bv[e] = (__bf16)V[bh + (size_t)ky * HID + j * 16 + col];
      }
      acc[j] = __builtin_amdgcn_wmma_f32_16x16x32_bf16(false, ap, false, bv, (short)0, acc[j], false, false);
    }
  }
  #pragma unroll
  for (int r = 0; r < 8; ++r) {
    const float inv = 1.0f / l_r[r];
    const size_t orow = ((size_t)b * S_LEN + q0 + r + hi * 8) * HID + (size_t)h * DHEAD + col;
    #pragma unroll
    for (int j = 0; j < 4; ++j) out[orow + j * 16] = acc[j][r] * inv;
  }
}

extern "C" void kernel_launch(void* const* d_in, const int* in_sizes, int n_in,
                              void* d_out, int out_size, void* d_ws, size_t ws_size,
                              hipStream_t stream) {
  const float* Q = (const float*)d_in[0];
  const float* K = (const float*)d_in[1];
  const float* V = (const float*)d_in[2];
  float* O = (float*)d_out;
  const int B  = in_sizes[0] / (S_LEN * HID);
  const int BH = B * NHEADS;
  const int waves = B * NHEADS * (S_LEN / 16);
  const size_t kpBytes = (size_t)BH * SP * DHEAD * 2;     // bf16
  const size_t needed  = 2 * kpBytes;

  if (ws_size >= needed) {
    __bf16* Kp = (__bf16*)d_ws;
    __bf16* Vp = (__bf16*)((char*)d_ws + kpBytes);
    const int nZero = BH * 2 * (WIN * DHEAD / 8) + BH * DHEAD * 2 * (WIN / 8);
    hipLaunchKernelGGL(zero_pads, dim3((nZero + 255) / 256), dim3(256), 0, stream, Kp, Vp, BH);
    hipLaunchKernelGGL(pack_kv, dim3(BH * (S_LEN / 64)), dim3(256), 0, stream, K, V, Kp, Vp);
    hipLaunchKernelGGL(swa_wmma_main, dim3(waves / 8), dim3(256), 0, stream, Q, Kp, Vp, O);
  } else {
    hipLaunchKernelGGL(swa_wmma_fallback, dim3(waves / 8), dim3(256), 0, stream, Q, K, V, O);
  }
}
